// Encoder_36146444763759
// MI455X (gfx1250) — compile-verified, matrix operand order checked by reading, not run on hardware
//
#include <hip/hip_runtime.h>
#include <hip/hip_bf16.h>

#define EMBED 1024
#define HEADS 16
#define HD 64
#define FFN 4096
#define SEQ 2048
#define BATCH 2
#define INPUT_DIM 64
#define NTOK (BATCH * SEQ)          // 4096 rows
#define NROWQ (NTOK * HEADS)        // 65536 rows for per-head projections

typedef __attribute__((ext_vector_type(16))) __bf16 v16bf;
typedef __attribute__((ext_vector_type(8)))  __bf16 v8bf;
typedef __attribute__((ext_vector_type(8)))  float  v8f;
typedef __attribute__((ext_vector_type(4)))  float  v4f;
typedef __attribute__((ext_vector_type(4)))  unsigned int ui4;
typedef __attribute__((ext_vector_type(8)))  int    i8i;
typedef __attribute__((ext_vector_type(4)))  int    i4i;

static __device__ __forceinline__ __bf16 f2bf(float f) {
    union { float f; unsigned u; } v; v.f = f;
    unsigned r = (v.u + 0x7FFFu + ((v.u >> 16) & 1u)) >> 16;
    union { unsigned short s; __bf16 b; } o; o.s = (unsigned short)r;
    return o.b;
}

// ---------------------------------------------------------------------------
// Tiled bf16-WMMA GEMM:  C[M,N] = act(A[M,K] @ B[K,N] + bias [+ pe])
// Block tile 64x64, K-step 32, 128 threads = 4 waves, wave owns 16 rows x 64 cols.
// A tile is DMA'd into LDS by the Tensor Data Mover (fp32, strided 2D tile);
// B tile is VALU-staged fp32->bf16 transposed (TDM cannot transpose).
// ---------------------------------------------------------------------------
#define BM 64
#define BN 64
#define BK 32

__global__ __launch_bounds__(128) void gemm_kernel(
    const float* __restrict__ A, const float* __restrict__ B,
    float* __restrict__ C, const float* __restrict__ bias,
    const float* __restrict__ pe, int M, int N, int K, int relu)
{
    __shared__ __align__(16) float  Asf[BM][BK];      // 8 KB, written by TDM
    __shared__ __align__(16) __bf16 Bs[BN][BK + 8];   // transposed [n][k]

    const int tid  = threadIdx.x;
    const int wave = tid >> 5;
    const int lane = tid & 31;
    const int g    = lane >> 4;     // half-wave group (K-chunk selector)
    const int l16  = lane & 15;
    const int rowBase = blockIdx.y * BM;
    const int colBase = blockIdx.x * BN;
    const unsigned ldsA = (unsigned)(size_t)(void*)Asf;  // LDS byte offset

    v8f acc[4];
    #pragma unroll
    for (int t = 0; t < 4; ++t)
        #pragma unroll
        for (int i = 0; i < 8; ++i) acc[t][i] = 0.0f;

    for (int k0 = 0; k0 < K; k0 += BK) {
        // ---- TDM: DMA the 64x32 fp32 A tile (row stride K) into LDS ------
        if (wave == 0) {
            const unsigned long long ga =
                (unsigned long long)(const void*)&A[(size_t)rowBase * K + k0];
            ui4 g0; i8i g1; i4i z4;
            g0[0] = 1u;                                   // count=1, user D#
            g0[1] = ldsA;                                 // lds_addr
            g0[2] = (unsigned)ga;                         // global_addr[31:0]
            g0[3] = (unsigned)((ga >> 32) & 0x01FFFFFFu)  // global_addr[56:32]
                  | (2u << 30);                           // type=2 ("image")
            g1[0] = (int)(2u << 16);                      // data_size=2 (4B)
            g1[1] = (int)(((unsigned)K & 0xFFFFu) << 16); // tensor_dim0 lo16
            g1[2] = (int)((((unsigned)K >> 16) & 0xFFFFu)
                  | (((unsigned)M & 0xFFFFu) << 16));     // dim0 hi / dim1 lo
            g1[3] = (int)((((unsigned)M >> 16) & 0xFFFFu)
                  | ((unsigned)BK << 16));                // dim1 hi / tile_dim0=32
            g1[4] = (int)(unsigned)BM;                    // tile_dim1=64, tile_dim2=0
            g1[5] = (int)(unsigned)K;                     // tensor_dim0_stride lo32
            g1[6] = 0;                                    // stride0 hi / stride1 lo
            g1[7] = 0;
            z4[0] = z4[1] = z4[2] = z4[3] = 0;
#if defined(__clang_major__) && (__clang_major__ >= 23)
            i8i z8;
            #pragma unroll
            for (int i = 0; i < 8; ++i) z8[i] = 0;
            __builtin_amdgcn_tensor_load_to_lds(g0, g1, z4, z4, z8, 0);
#else
            __builtin_amdgcn_tensor_load_to_lds(g0, g1, z4, z4, 0);
#endif
        }

        // ---- stage B tile transposed (32x64 fp32 -> Bs[n][k] bf16) -------
        for (int idx = tid; idx < BK * BN; idx += 128) {
            int r = idx >> 6, c = idx & 63;
            Bs[c][r] = f2bf(B[(size_t)(k0 + r) * N + colBase + c]);
        }

        if (k0 + BK < K)
            __builtin_prefetch(&B[(size_t)(k0 + BK) * N + colBase], 0, 1);

        if (wave == 0) __builtin_amdgcn_s_wait_tensorcnt(0);
        __syncthreads();

        // A fragment: 16-bit A 16x32 layout (ISA 7.12.2): lane m=l16,
        // bf16 elems 0..7 at K=g*8.., elems 8..15 at K=16+g*8..
        const int am = wave * 16 + l16;
        v4f a0 = *(const v4f*)&Asf[am][g * 8];
        v4f a1 = *(const v4f*)&Asf[am][g * 8 + 4];
        v4f a2 = *(const v4f*)&Asf[am][16 + g * 8];
        v4f a3 = *(const v4f*)&Asf[am][16 + g * 8 + 4];
        v16bf afrag;
        #pragma unroll
        for (int i = 0; i < 4; ++i) {
            afrag[i]      = f2bf(a0[i]);
            afrag[i + 4]  = f2bf(a1[i]);
            afrag[i + 8]  = f2bf(a2[i]);
            afrag[i + 12] = f2bf(a3[i]);
        }

        #pragma unroll
        for (int t = 0; t < 4; ++t) {
            const int bn = t * 16 + l16;      // column index of B
            v8bf blo = *(const v8bf*)&Bs[bn][g * 8];
            v8bf bhi = *(const v8bf*)&Bs[bn][16 + g * 8];
            v16bf bfrag;
            #pragma unroll
            for (int i = 0; i < 8; ++i) { bfrag[i] = blo[i]; bfrag[i + 8] = bhi[i]; }
            acc[t] = __builtin_amdgcn_wmma_f32_16x16x32_bf16(
                false, afrag, false, bfrag, (short)0, acc[t], false, false);
        }
        __syncthreads();
    }

    // epilogue: C layout (7.12.2): VGPR r -> row g*8+r, lane%16 -> col
    const int orow = rowBase + wave * 16 + g * 8;
    #pragma unroll
    for (int t = 0; t < 4; ++t) {
        const int ocol = colBase + t * 16 + l16;
        const float bv = bias ? bias[ocol] : 0.0f;
        #pragma unroll
        for (int r = 0; r < 8; ++r) {
            float v = acc[t][r] + bv;
            if (pe) v += pe[(size_t)((orow + r) % SEQ) * N + ocol];
            if (relu) v = fmaxf(v, 0.0f);
            C[(size_t)(orow + r) * N + ocol] = v;
        }
    }
}

// ---------------------------------------------------------------------------
// Flash attention per (batch, head, 64-query block).  128 thr = 4 waves,
// each wave owns 16 query rows.  Online softmax, P staged to per-wave LDS.
// ---------------------------------------------------------------------------
__global__ __launch_bounds__(128) void attn_kernel(
    const float* __restrict__ Q, const float* __restrict__ K,
    const float* __restrict__ V, const unsigned char* __restrict__ mask,
    float* __restrict__ O, float scale)
{
    __shared__ __align__(16) __bf16 Qs[64][HD];        // 8 KB
    __shared__ __align__(16) __bf16 Ks[64][HD];        // 8 KB
    __shared__ __align__(16) __bf16 Vt[HD][64];        // 8 KB, transposed [d][k]
    __shared__ __align__(16) __bf16 Ps[4][16][64];     // 8 KB, per-wave P tile

    const int qb = blockIdx.x, hh = blockIdx.y, b = blockIdx.z;
    const int tid  = threadIdx.x;
    const int wave = tid >> 5;
    const int lane = tid & 31;
    const int g    = lane >> 4;
    const int l16  = lane & 15;

    // load Q block (64 rows x 64 dims), fp32 -> bf16
    for (int idx = tid; idx < 64 * HD; idx += 128) {
        int r = idx >> 6, c = idx & 63;
        int s = qb * 64 + r;
        Qs[r][c] = f2bf(Q[((size_t)(b * SEQ + s) * HEADS + hh) * HD + c]);
    }
    __syncthreads();

    // per-wave Q fragments (d 0..31 and d 32..63), reused across key loop
    const int qm = wave * 16 + l16;
    v16bf aq[2];
    #pragma unroll
    for (int h = 0; h < 2; ++h) {
        v8bf lo = *(const v8bf*)&Qs[qm][h * 32 + g * 8];
        v8bf hi = *(const v8bf*)&Qs[qm][h * 32 + 16 + g * 8];
        #pragma unroll
        for (int i = 0; i < 8; ++i) { aq[h][i] = lo[i]; aq[h][i + 8] = hi[i]; }
    }

    float m[8], ssum[8];
    v8f o[4];
    #pragma unroll
    for (int r = 0; r < 8; ++r) { m[r] = -1e30f; ssum[r] = 0.0f; }
    #pragma unroll
    for (int t = 0; t < 4; ++t)
        #pragma unroll
        for (int i = 0; i < 8; ++i) o[t][i] = 0.0f;

    for (int kb = 0; kb < SEQ; kb += 64) {
        __syncthreads();   // previous iteration's reads done before overwrite
        for (int idx = tid; idx < 64 * HD; idx += 128) {
            int r = idx >> 6, c = idx & 63;
            Ks[r][c] = f2bf(K[((size_t)(b * SEQ + kb + r) * HEADS + hh) * HD + c]);
            Vt[c][r] = f2bf(V[((size_t)(b * SEQ + kb + r) * HEADS + hh) * HD + c]);
        }
        __syncthreads();

        // S = Q @ K^T for 4 key sub-tiles (16 keys each)
        v8f st[4];
        #pragma unroll
        for (int t = 0; t < 4; ++t) {
            const int kn = t * 16 + l16;
            v16bf bk0, bk1;
            {
                v8bf lo = *(const v8bf*)&Ks[kn][g * 8];
                v8bf hi = *(const v8bf*)&Ks[kn][16 + g * 8];
                #pragma unroll
                for (int i = 0; i < 8; ++i) { bk0[i] = lo[i]; bk0[i + 8] = hi[i]; }
            }
            {
                v8bf lo = *(const v8bf*)&Ks[kn][32 + g * 8];
                v8bf hi = *(const v8bf*)&Ks[kn][48 + g * 8];
                #pragma unroll
                for (int i = 0; i < 8; ++i) { bk1[i] = lo[i]; bk1[i + 8] = hi[i]; }
            }
            v8f s;
            #pragma unroll
            for (int i = 0; i < 8; ++i) s[i] = 0.0f;
            s = __builtin_amdgcn_wmma_f32_16x16x32_bf16(false, aq[0], false, bk0, (short)0, s, false, false);
            s = __builtin_amdgcn_wmma_f32_16x16x32_bf16(false, aq[1], false, bk1, (short)0, s, false, false);
            const unsigned char mv = mask[b * SEQ + kb + kn];
            #pragma unroll
            for (int r = 0; r < 8; ++r) s[r] = mv ? s[r] * scale : -1e20f;
            st[t] = s;
        }

        // online softmax: row stats span 16 lanes within each half-wave
        #pragma unroll
        for (int r = 0; r < 8; ++r) {
            float lm = fmaxf(fmaxf(st[0][r], st[1][r]), fmaxf(st[2][r], st[3][r]));
            #pragma unroll
            for (int off = 1; off < 16; off <<= 1)
                lm = fmaxf(lm, __shfl_xor(lm, off, 32));
            const float mnew  = fmaxf(m[r], lm);
            const float alpha = __expf(m[r] - mnew);
            m[r] = mnew;
            float ps = 0.0f;
            #pragma unroll
            for (int t = 0; t < 4; ++t) {
                float p = __expf(st[t][r] - mnew);
                st[t][r] = p; ps += p;
            }
            #pragma unroll
            for (int off = 1; off < 16; off <<= 1)
                ps += __shfl_xor(ps, off, 32);
            ssum[r] = ssum[r] * alpha + ps;
            #pragma unroll
            for (int t = 0; t < 4; ++t) o[t][r] *= alpha;
        }

        // stage P (C-layout) -> per-wave LDS tile, re-read as A fragments
        #pragma unroll
        for (int t = 0; t < 4; ++t)
            #pragma unroll
            for (int r = 0; r < 8; ++r)
                Ps[wave][g * 8 + r][t * 16 + l16] = f2bf(st[t][r]);
        asm volatile("s_wait_dscnt 0" ::: "memory");   // own-wave DS RAW fence

        v16bf pa[2];
        #pragma unroll
        for (int h = 0; h < 2; ++h) {
            v8bf lo = *(const v8bf*)&Ps[wave][l16][h * 32 + g * 8];
            v8bf hi = *(const v8bf*)&Ps[wave][l16][h * 32 + 16 + g * 8];
            #pragma unroll
            for (int i = 0; i < 8; ++i) { pa[h][i] = lo[i]; pa[h][i + 8] = hi[i]; }
        }

        // O += P @ V  (Vt[d][k]: B columns are output dims, contiguous in k)
        #pragma unroll
        for (int t = 0; t < 4; ++t) {
            const int dn = t * 16 + l16;
            v16bf bv0, bv1;
            {
                v8bf lo = *(const v8bf*)&Vt[dn][g * 8];
                v8bf hi = *(const v8bf*)&Vt[dn][16 + g * 8];
                #pragma unroll
                for (int i = 0; i < 8; ++i) { bv0[i] = lo[i]; bv0[i + 8] = hi[i]; }
            }
            {
                v8bf lo = *(const v8bf*)&Vt[dn][32 + g * 8];
                v8bf hi = *(const v8bf*)&Vt[dn][48 + g * 8];
                #pragma unroll
                for (int i = 0; i < 8; ++i) { bv1[i] = lo[i]; bv1[i + 8] = hi[i]; }
            }
            o[t] = __builtin_amdgcn_wmma_f32_16x16x32_bf16(false, pa[0], false, bv0, (short)0, o[t], false, false);
            o[t] = __builtin_amdgcn_wmma_f32_16x16x32_bf16(false, pa[1], false, bv1, (short)0, o[t], false, false);
        }
    }

    // normalize and write out in [B,S,H,D] (contiguous == [B,S,E])
    #pragma unroll
    for (int t = 0; t < 4; ++t) {
        #pragma unroll
        for (int r = 0; r < 8; ++r) {
            const int q = qb * 64 + wave * 16 + g * 8 + r;
            O[((size_t)(b * SEQ + q) * HEADS + hh) * HD + t * 16 + l16] =
                o[t][r] / ssum[r];
        }
    }
}

// ---------------------------------------------------------------------------
// Fused residual-add + LayerNorm: out[row] = LN(a[row] + res[row]) * g + b
// One 256-thread block per row of 1024 columns.
// ---------------------------------------------------------------------------
__global__ __launch_bounds__(256) void ln_kernel(
    const float* __restrict__ a, const float* __restrict__ res,
    const float* __restrict__ gam, const float* __restrict__ bet,
    float* __restrict__ out)
{
    __shared__ float red[256];
    const int row = blockIdx.x;
    const int tid = threadIdx.x;
    const size_t base = (size_t)row * EMBED;

    float v[4];
    float s = 0.0f;
    #pragma unroll
    for (int i = 0; i < 4; ++i) {
        const int c = tid + i * 256;
        v[i] = a[base + c] + res[base + c];
        s += v[i];
    }
    red[tid] = s; __syncthreads();
    for (int off = 128; off > 0; off >>= 1) {
        if (tid < off) red[tid] += red[tid + off];
        __syncthreads();
    }
    const float mu = red[0] * (1.0f / EMBED);
    __syncthreads();

    float s2 = 0.0f;
    #pragma unroll
    for (int i = 0; i < 4; ++i) { float d = v[i] - mu; s2 += d * d; }
    red[tid] = s2; __syncthreads();
    for (int off = 128; off > 0; off >>= 1) {
        if (tid < off) red[tid] += red[tid + off];
        __syncthreads();
    }
    const float rs = rsqrtf(red[0] * (1.0f / EMBED) + 1e-5f);
    #pragma unroll
    for (int i = 0; i < 4; ++i) {
        const int c = tid + i * 256;
        out[base + c] = (v[i] - mu) * rs * gam[c] + bet[c];
    }
}

// ---------------------------------------------------------------------------
extern "C" void kernel_launch(void* const* d_in, const int* in_sizes, int n_in,
                              void* d_out, int out_size, void* d_ws, size_t ws_size,
                              hipStream_t stream) {
    (void)in_sizes; (void)n_in; (void)out_size; (void)ws_size;
    const float*         x       = (const float*)d_in[0];
    const unsigned char* mask    = (const unsigned char*)d_in[1];
    const float*         embed_W = (const float*)d_in[2];
    const float*         embed_b = (const float*)d_in[3];
    const float*         pe      = (const float*)d_in[4];
    const float*         Wq      = (const float*)d_in[5];
    const float*         Wk      = (const float*)d_in[6];
    const float*         Wv      = (const float*)d_in[7];
    const float*         Wo      = (const float*)d_in[8];
    const float*         bo      = (const float*)d_in[9];
    const float*         ln1_g   = (const float*)d_in[10];
    const float*         ln1_b   = (const float*)d_in[11];
    const float*         W1      = (const float*)d_in[12];
    const float*         b1      = (const float*)d_in[13];
    const float*         W2      = (const float*)d_in[14];
    const float*         b2      = (const float*)d_in[15];
    const float*         ln2_g   = (const float*)d_in[16];
    const float*         ln2_b   = (const float*)d_in[17];

    float* h  = (float*)d_out;                 // [B,S,E] master activation
    float* ws = (float*)d_ws;
    const size_t TOKE = (size_t)NTOK * EMBED;  // 4 M floats
    float* Qb  = ws;
    float* Kb  = Qb  + TOKE;
    float* Vb  = Kb  + TOKE;
    float* ATT = Vb  + TOKE;
    float* OPJ = ATT + TOKE;
    float* X1  = OPJ + TOKE;
    float* FF1 = X1  + TOKE;                   // [4096,4096] = 16 M floats

    const dim3 blk(128);
    const float scale = 1.0f / 32.0f;          // 1/sqrt(EMBED)

    // h = x @ embed_W + embed_b + pe
    gemm_kernel<<<dim3(EMBED / BN, NTOK / BM), blk, 0, stream>>>(
        x, embed_W, h, embed_b, pe, NTOK, EMBED, INPUT_DIM, 0);

    for (int l = 0; l < 3; ++l) {
        const float* Wql = Wq + (size_t)l * HD * HD;
        const float* Wkl = Wk + (size_t)l * HD * HD;
        const float* Wvl = Wv + (size_t)l * HD * HD;
        const float* Wol = Wo + (size_t)l * EMBED * EMBED;
        const float* W1l = W1 + (size_t)l * EMBED * FFN;
        const float* W2l = W2 + (size_t)l * FFN * EMBED;

        // per-head QKV projections == plain GEMMs [65536,64]@[64,64]
        gemm_kernel<<<dim3(1, NROWQ / BM), blk, 0, stream>>>(h, Wql, Qb, nullptr, nullptr, NROWQ, HD, HD, 0);
        gemm_kernel<<<dim3(1, NROWQ / BM), blk, 0, stream>>>(h, Wkl, Kb, nullptr, nullptr, NROWQ, HD, HD, 0);
        gemm_kernel<<<dim3(1, NROWQ / BM), blk, 0, stream>>>(h, Wvl, Vb, nullptr, nullptr, NROWQ, HD, HD, 0);

        attn_kernel<<<dim3(SEQ / 64, HEADS, BATCH), blk, 0, stream>>>(
            Qb, Kb, Vb, mask, ATT, scale);

        // o = attn @ Wo + bo
        gemm_kernel<<<dim3(EMBED / BN, NTOK / BM), blk, 0, stream>>>(
            ATT, Wol, OPJ, bo + (size_t)l * EMBED, nullptr, NTOK, EMBED, EMBED, 0);
        // x1 = LN(o + h)
        ln_kernel<<<NTOK, 256, 0, stream>>>(OPJ, h,
            ln1_g + (size_t)l * EMBED, ln1_b + (size_t)l * EMBED, X1);
        // ff1 = relu(x1 @ W1 + b1)
        gemm_kernel<<<dim3(FFN / BN, NTOK / BM), blk, 0, stream>>>(
            X1, W1l, FF1, b1 + (size_t)l * FFN, nullptr, NTOK, FFN, EMBED, 1);
        // ff2 = ff1 @ W2 + b2
        gemm_kernel<<<dim3(EMBED / BN, NTOK / BM), blk, 0, stream>>>(
            FF1, W2l, OPJ, b2 + (size_t)l * EMBED, nullptr, NTOK, EMBED, FFN, 0);
        // h = LN(ff2 + x1)
        ln_kernel<<<NTOK, 256, 0, stream>>>(OPJ, X1,
            ln2_g + (size_t)l * EMBED, ln2_b + (size_t)l * EMBED, h);
    }
}
